// FineAMT_7258494730456
// MI455X (gfx1250) — matrix operand order repaired and checked by reading, not compiled, not run on hardware
//
#include <hip/hip_runtime.h>
#include <hip/hip_bf16.h>

typedef __attribute__((ext_vector_type(16))) _Float16 v16h;
typedef __attribute__((ext_vector_type(8)))  float    v8f;
typedef __attribute__((ext_vector_type(4)))  float    f32x4;
typedef __attribute__((ext_vector_type(4)))  _Float16 f16x4;

#define BSZ     4
#define SEQL    2048
#define DMODEL  512
#define D1C     128
#define D2C     384
#define DSTATE  64
#define DCONV   4
#define DINNER  1024
#define HEADDIM 64
#define HSSM    16
#define NHEAD   8
#define NEXP    8
#define FFN     2048
#define NPITCH  128
#define CONVDIM (DINNER + 2*DSTATE)          /* 1152 */
#define DIP     (2*DINNER + 2*DSTATE + HSSM) /* 2192 */
#define NTOK    (BSZ*SEQL)                   /* 8192 */
#define EPS_RMS 1.1920929e-07f

__device__ __forceinline__ float act_apply(float v, int act) {
  switch (act) {
    case 1: return 0.5f * v * (1.0f + erff(v * 0.70710678118654752f)); // exact gelu
    case 2: return v / (1.0f + __expf(-v));                           // silu
    case 3: return 1.0f / (1.0f + __expf(-v));                        // sigmoid
    default: return v;
  }
}

// ---------------------------------------------------------------------------
// WMMA GEMM: C[M,N] = act(A[M,K] * W[N,K]^T + bias) * row_scale (+=)
// Requirements (all call sites satisfy): K % 32 == 0, M covered by grid
// (M % 128 may have a ragged last block -> row-guarded), N ragged -> guarded.
// Block: 256 thr = 8 waves, 128x64 tile, each wave a 32x32 sub-tile
// (4 WMMAs / K-step). Double-buffered LDS, vector (b128) staging.
// ---------------------------------------------------------------------------
__global__ __launch_bounds__(256)
void gemm_wmma_kernel(const float* __restrict__ A, int lda,
                      const float* __restrict__ W,
                      const float* __restrict__ bias,
                      const float* __restrict__ rscale, int rs_stride,
                      float* __restrict__ C, int ldc,
                      int M, int N, int K, int act, int accum)
{
  __shared__ _Float16 As[2][128][40];   // 40-half rows: 80B = 5x16B aligned
  __shared__ _Float16 Bs[2][64][40];
  const int tid  = threadIdx.x;
  const int lane = tid & 31;
  const int wave = tid >> 5;
  const int half = lane >> 4, l16 = lane & 15;
  const int rowBase = blockIdx.y * 128;
  const int colBase = blockIdx.x * 64;
  const int wm = (wave & 3) * 32;
  const int wn = (wave >> 2) * 32;

  // staging assignment: A: thread -> (row tid/2, 16 k's); W: (row tid/4, 8 k's)
  const int ar  = tid >> 1;
  const int ac  = (tid & 1) * 16;
  const int wr  = tid >> 2;
  const int wc  = (tid & 3) * 8;
  const int agr = rowBase + ar;
  const int wgn = colBase + wr;
  const bool a_ok = (agr < M);
  const bool w_ok = (wgn < N);

  f16x4 ah[4], wh[2];

  auto stage = [&](int k0) {
    const float* ap = A + (size_t)agr * lda + k0 + ac;
    if (a_ok) {
      #pragma unroll
      for (int j = 0; j < 4; ++j)
        ah[j] = __builtin_convertvector(*(const f32x4*)(ap + 4 * j), f16x4);
    } else {
      #pragma unroll
      for (int j = 0; j < 4; ++j) ah[j] = (f16x4){};
    }
    const float* wp = W + (size_t)wgn * K + k0 + wc;
    if (w_ok) {
      #pragma unroll
      for (int j = 0; j < 2; ++j)
        wh[j] = __builtin_convertvector(*(const f32x4*)(wp + 4 * j), f16x4);
    } else {
      #pragma unroll
      for (int j = 0; j < 2; ++j) wh[j] = (f16x4){};
    }
    if (k0 + 64 < K) {   // CDNA5 global_prefetch_b8 for the tile after next
      if (a_ok) __builtin_prefetch(ap + 32, 0, 1);
      if (w_ok) __builtin_prefetch(wp + 32, 0, 1);
    }
  };
  auto commit = [&](int buf) {
    #pragma unroll
    for (int j = 0; j < 4; ++j) *(f16x4*)&As[buf][ar][ac + 4 * j] = ah[j];
    #pragma unroll
    for (int j = 0; j < 2; ++j) *(f16x4*)&Bs[buf][wr][wc + 4 * j] = wh[j];
  };

  v8f acc00 = {}, acc01 = {}, acc10 = {}, acc11 = {};

  stage(0);
  commit(0);
  int buf = 0;
  for (int k0 = 0; k0 < K; k0 += 32) {
    __syncthreads();
    const bool more = (k0 + 32 < K);
    if (more) stage(k0 + 32);          // global loads in flight during WMMA

    // fragments (contiguous k = 16*half + s in both A and B, wave32)
    v16h a0, a1, b0, b1;
    const int ka = half * 16;
    #pragma unroll
    for (int v = 0; v < 4; ++v) {
      a0[2*v]     = As[buf][wm + l16][ka + 2*v];
      a0[2*v + 1] = As[buf][wm + l16][ka + 2*v + 1];
      a0[2*v + 8] = As[buf][wm + l16][8 + ka + 2*v];
      a0[2*v + 9] = As[buf][wm + l16][8 + ka + 2*v + 1];
      a1[2*v]     = As[buf][wm + 16 + l16][ka + 2*v];
      a1[2*v + 1] = As[buf][wm + 16 + l16][ka + 2*v + 1];
      a1[2*v + 8] = As[buf][wm + 16 + l16][8 + ka + 2*v];
      a1[2*v + 9] = As[buf][wm + 16 + l16][8 + ka + 2*v + 1];
    }
    const int kb = half * 16;
    #pragma unroll
    for (int v = 0; v < 8; ++v) {
      b0[2*v]     = Bs[buf][wn + l16][kb + 2*v];
      b0[2*v + 1] = Bs[buf][wn + l16][kb + 2*v + 1];
      b1[2*v]     = Bs[buf][wn + 16 + l16][kb + 2*v];
      b1[2*v + 1] = Bs[buf][wn + 16 + l16][kb + 2*v + 1];
    }
    acc00 = __builtin_amdgcn_wmma_f32_16x16x32_f16(false, a0, false, b0, (short)0, acc00, false, false);
    acc01 = __builtin_amdgcn_wmma_f32_16x16x32_f16(false, a0, false, b1, (short)0, acc01, false, false);
    acc10 = __builtin_amdgcn_wmma_f32_16x16x32_f16(false, a1, false, b0, (short)0, acc10, false, false);
    acc11 = __builtin_amdgcn_wmma_f32_16x16x32_f16(false, a1, false, b1, (short)0, acc11, false, false);

    if (more) commit(buf ^ 1);
    buf ^= 1;
  }

  // epilogue: VGPR d holds M = d (+8 for lanes 16-31), N = lane%16
  const int gc0 = colBase + wn + l16;
  const int gc1 = gc0 + 16;
  #pragma unroll
  for (int d = 0; d < 8; ++d) {
    const int r0 = rowBase + wm + d + half * 8;
    const int r1 = r0 + 16;
    if (r0 < M) {
      const float rs = rscale ? rscale[(size_t)r0 * rs_stride] : 1.0f;
      if (gc0 < N) {
        float v = act_apply(acc00[d] + (bias ? bias[gc0] : 0.0f), act) * rs;
        size_t off = (size_t)r0 * ldc + gc0;
        C[off] = accum ? (C[off] + v) : v;
      }
      if (gc1 < N) {
        float v = act_apply(acc01[d] + (bias ? bias[gc1] : 0.0f), act) * rs;
        size_t off = (size_t)r0 * ldc + gc1;
        C[off] = accum ? (C[off] + v) : v;
      }
    }
    if (r1 < M) {
      const float rs = rscale ? rscale[(size_t)r1 * rs_stride] : 1.0f;
      if (gc0 < N) {
        float v = act_apply(acc10[d] + (bias ? bias[gc0] : 0.0f), act) * rs;
        size_t off = (size_t)r1 * ldc + gc0;
        C[off] = accum ? (C[off] + v) : v;
      }
      if (gc1 < N) {
        float v = act_apply(acc11[d] + (bias ? bias[gc1] : 0.0f), act) * rs;
        size_t off = (size_t)r1 * ldc + gc1;
        C[off] = accum ? (C[off] + v) : v;
      }
    }
  }
}

// ---------------------------------------------------------------------------
// Embedding combine + LayerNorm (D=512), pad rows zeroed. One block per token.
// ---------------------------------------------------------------------------
__global__ __launch_bounds__(256)
void embed_ln_kernel(const float* __restrict__ p1, const float* __restrict__ p2,
                     const int* __restrict__ tids,
                     const float* __restrict__ type_emb, const float* __restrict__ pos_emb,
                     const float* __restrict__ lnw, const float* __restrict__ lnb,
                     float* __restrict__ out)
{
  const int row = blockIdx.x;
  const int l   = row % SEQL;
  const int tv  = tids[row];
  __shared__ float red[256];
  __shared__ float s_mean, s_rstd;
  float vals[2];
  float lsum = 0.f;
  #pragma unroll
  for (int i = 0; i < 2; ++i) {
    const int d = threadIdx.x + i * 256;
    float v = 0.f;
    if (tv == 0)      v = p1[(size_t)row * DMODEL + d];
    else if (tv == 1) v = p2[(size_t)row * DMODEL + d];
    const int tc = tv < 0 ? 0 : tv;
    v += type_emb[tc * DMODEL + d] + pos_emb[(size_t)l * DMODEL + d];
    vals[i] = v; lsum += v;
  }
  red[threadIdx.x] = lsum; __syncthreads();
  for (int s = 128; s > 0; s >>= 1) { if (threadIdx.x < s) red[threadIdx.x] += red[threadIdx.x + s]; __syncthreads(); }
  if (threadIdx.x == 0) s_mean = red[0] * (1.f / 512.f);
  __syncthreads();
  const float mean = s_mean;
  float lvar = 0.f;
  #pragma unroll
  for (int i = 0; i < 2; ++i) { float dv = vals[i] - mean; lvar += dv * dv; }
  __syncthreads();
  red[threadIdx.x] = lvar; __syncthreads();
  for (int s = 128; s > 0; s >>= 1) { if (threadIdx.x < s) red[threadIdx.x] += red[threadIdx.x + s]; __syncthreads(); }
  if (threadIdx.x == 0) s_rstd = rsqrtf(red[0] * (1.f / 512.f) + 1e-5f);
  __syncthreads();
  const float rstd = s_rstd;
  #pragma unroll
  for (int i = 0; i < 2; ++i) {
    const int d = threadIdx.x + i * 256;
    float v = (tv == -1) ? 0.f : ((vals[i] - mean) * rstd * lnw[d] + lnb[d]);
    out[(size_t)row * DMODEL + d] = v;
  }
}

// Generic LayerNorm over D=512
__global__ __launch_bounds__(256)
void ln512_kernel(const float* __restrict__ in, const float* __restrict__ w,
                  const float* __restrict__ b, float* __restrict__ out)
{
  const int row = blockIdx.x;
  __shared__ float red[256];
  __shared__ float s_mean, s_rstd;
  float vals[2]; float lsum = 0.f;
  #pragma unroll
  for (int i = 0; i < 2; ++i) {
    const int d = threadIdx.x + i * 256;
    vals[i] = in[(size_t)row * DMODEL + d]; lsum += vals[i];
  }
  red[threadIdx.x] = lsum; __syncthreads();
  for (int s = 128; s > 0; s >>= 1) { if (threadIdx.x < s) red[threadIdx.x] += red[threadIdx.x + s]; __syncthreads(); }
  if (threadIdx.x == 0) s_mean = red[0] * (1.f / 512.f);
  __syncthreads();
  const float mean = s_mean;
  float lvar = 0.f;
  #pragma unroll
  for (int i = 0; i < 2; ++i) { float dv = vals[i] - mean; lvar += dv * dv; }
  __syncthreads();
  red[threadIdx.x] = lvar; __syncthreads();
  for (int s = 128; s > 0; s >>= 1) { if (threadIdx.x < s) red[threadIdx.x] += red[threadIdx.x + s]; __syncthreads(); }
  if (threadIdx.x == 0) s_rstd = rsqrtf(red[0] * (1.f / 512.f) + 1e-5f);
  __syncthreads();
  const float rstd = s_rstd;
  #pragma unroll
  for (int i = 0; i < 2; ++i) {
    const int d = threadIdx.x + i * 256;
    out[(size_t)row * DMODEL + d] = (vals[i] - mean) * rstd * w[d] + b[d];
  }
}

// RMSNorm over D=512
__global__ __launch_bounds__(256)
void rms512_kernel(const float* __restrict__ in, const float* __restrict__ w,
                   float* __restrict__ out)
{
  const int row = blockIdx.x;
  __shared__ float red[256];
  __shared__ float s_inv;
  float vals[2]; float lsum = 0.f;
  #pragma unroll
  for (int i = 0; i < 2; ++i) {
    const int d = threadIdx.x + i * 256;
    vals[i] = in[(size_t)row * DMODEL + d]; lsum += vals[i] * vals[i];
  }
  red[threadIdx.x] = lsum; __syncthreads();
  for (int s = 128; s > 0; s >>= 1) { if (threadIdx.x < s) red[threadIdx.x] += red[threadIdx.x + s]; __syncthreads(); }
  if (threadIdx.x == 0) s_inv = rsqrtf(red[0] * (1.f / 512.f) + EPS_RMS);
  __syncthreads();
  const float inv = s_inv;
  #pragma unroll
  for (int i = 0; i < 2; ++i) {
    const int d = threadIdx.x + i * 256;
    out[(size_t)row * DMODEL + d] = vals[i] * inv * w[d];
  }
}

// Gated RMSNorm over DINNER=1024: out = rms(y * silu(z)) * norm_w
__global__ __launch_bounds__(256)
void gated_rms_kernel(const float* __restrict__ y, const float* __restrict__ zx,
                      const float* __restrict__ nw, float* __restrict__ out)
{
  const int row = blockIdx.x;
  __shared__ float red[256];
  __shared__ float s_inv;
  float u[4]; float lsum = 0.f;
  #pragma unroll
  for (int i = 0; i < 4; ++i) {
    const int d = threadIdx.x + i * 256;
    const float z = zx[(size_t)row * DIP + d];
    const float v = y[(size_t)row * DINNER + d] * (z / (1.f + __expf(-z)));
    u[i] = v; lsum += v * v;
  }
  red[threadIdx.x] = lsum; __syncthreads();
  for (int s = 128; s > 0; s >>= 1) { if (threadIdx.x < s) red[threadIdx.x] += red[threadIdx.x + s]; __syncthreads(); }
  if (threadIdx.x == 0) s_inv = rsqrtf(red[0] * (1.f / 1024.f) + 1e-5f);
  __syncthreads();
  const float inv = s_inv;
  #pragma unroll
  for (int i = 0; i < 4; ++i) {
    const int d = threadIdx.x + i * 256;
    out[(size_t)row * DINNER + d] = u[i] * inv * nw[d];
  }
}

// Depthwise causal conv (width 4) + SiLU on xBC slice of zxbcdt
__global__ __launch_bounds__(256)
void conv_silu_kernel(const float* __restrict__ zx, const float* __restrict__ cw,
                      const float* __restrict__ cb, float* __restrict__ out)
{
  const size_t i = (size_t)blockIdx.x * 256 + threadIdx.x;
  if (i >= (size_t)NTOK * CONVDIM) return;
  const int c = (int)(i % CONVDIM);
  const size_t row = i / CONVDIM;
  const int l = (int)(row % SEQL);
  float acc = cb[c];
  #pragma unroll
  for (int k = 0; k < DCONV; ++k) {
    const int ls = l - (DCONV - 1) + k;
    if (ls >= 0)
      acc += cw[c * DCONV + k] * zx[(row - (size_t)(l - ls)) * DIP + DINNER + c];
  }
  out[row * CONVDIM + c] = acc / (1.f + __expf(-acc));
}

// dt = softplus(dtr + dt_bias); dA = exp(-exp(A_log) * dt)
__global__ __launch_bounds__(256)
void dt_kernel(const float* __restrict__ zx, const float* __restrict__ dt_bias,
               const float* __restrict__ A_log, float* __restrict__ dtb,
               float* __restrict__ dab)
{
  const int i = blockIdx.x * 256 + threadIdx.x;
  if (i >= NTOK * HSSM) return;
  const int hh = i % HSSM;
  const size_t row = (size_t)(i / HSSM);
  const float xv = zx[row * DIP + (DIP - HSSM) + hh] + dt_bias[hh];
  const float dt = (xv > 20.f) ? xv : log1pf(__expf(xv));
  dtb[i] = dt;
  dab[i] = __expf(-__expf(A_log[hh]) * dt);
}

// Sequential SSM scan: one block per (batch, ssm-head); 256 threads own the
// 64x64 state (16 elems each). y[b,t,h*64+p] = sum_n h[p,n]*C[n] + Dp*x[p].
__global__ __launch_bounds__(256)
void scan_kernel(const float* __restrict__ conv, const float* __restrict__ dtb,
                 const float* __restrict__ dab, const float* __restrict__ Dp,
                 float* __restrict__ y)
{
  const int b  = blockIdx.x >> 4;
  const int hh = blockIdx.x & 15;
  const int tid = threadIdx.x;
  const int p  = tid >> 2;
  const int nq = (tid & 3) << 4;
  float h[16];
  #pragma unroll
  for (int j = 0; j < 16; ++j) h[j] = 0.f;
  __shared__ float xsh[64], Bsh[64], Csh[64];
  __shared__ float red[256];
  const float dpv = Dp[hh];
  for (int t = 0; t < SEQL; ++t) {
    const size_t row = (size_t)b * SEQL + t;
    if (tid < 64)        xsh[tid]       = conv[row * CONVDIM + hh * HEADDIM + tid];
    else if (tid < 128)  Bsh[tid - 64]  = conv[row * CONVDIM + DINNER + (tid - 64)];
    else if (tid < 192)  Csh[tid - 128] = conv[row * CONVDIM + DINNER + DSTATE + (tid - 128)];
    __syncthreads();
    const float dA = dab[row * HSSM + hh];
    const float dt = dtb[row * HSSM + hh];
    const float xp = dt * xsh[p];
    float acc = 0.f;
    #pragma unroll
    for (int j = 0; j < 16; ++j) {
      h[j] = dA * h[j] + xp * Bsh[nq + j];
      acc += h[j] * Csh[nq + j];
    }
    red[tid] = acc;
    __syncthreads();
    if ((tid & 3) == 0) {
      const float yv = red[tid] + red[tid + 1] + red[tid + 2] + red[tid + 3] + dpv * xsh[p];
      y[row * DINNER + hh * HEADDIM + p] = yv;
    }
    __syncthreads();
  }
}

// Attention softmax: one block per (q, head, batch). qkv rows are 1536 wide.
__global__ __launch_bounds__(256)
void attn_kernel(const float* __restrict__ qkv, const int* __restrict__ tids,
                 float* __restrict__ o)
{
  const int qi = blockIdx.x, hh = blockIdx.y, b = blockIdx.z;
  const int tid = threadIdx.x;
  __shared__ float sc[SEQL];
  __shared__ float qs[HEADDIM];
  __shared__ float red[256];
  const size_t base = (size_t)b * SEQL * 1536;
  if (tid < 64) qs[tid] = qkv[base + (size_t)qi * 1536 + hh * 64 + tid];
  __syncthreads();
  float lmax = -1e30f;
  for (int k = tid; k < SEQL; k += 256) {
    const float* kv = &qkv[base + (size_t)k * 1536 + 512 + hh * 64];
    float d = 0.f;
    #pragma unroll
    for (int j = 0; j < 64; ++j) d += qs[j] * kv[j];
    d *= 0.125f;
    if (tids[b * SEQL + k] == -1) d = -1e9f;
    sc[k] = d;
    lmax = fmaxf(lmax, d);
  }
  red[tid] = lmax; __syncthreads();
  for (int s = 128; s > 0; s >>= 1) { if (tid < s) red[tid] = fmaxf(red[tid], red[tid + s]); __syncthreads(); }
  const float mx = red[0];
  __syncthreads();
  float lsum = 0.f;
  for (int k = tid; k < SEQL; k += 256) { const float e = __expf(sc[k] - mx); sc[k] = e; lsum += e; }
  red[tid] = lsum; __syncthreads();
  for (int s = 128; s > 0; s >>= 1) { if (tid < s) red[tid] += red[tid + s]; __syncthreads(); }
  const float inv = 1.f / red[0];
  __syncthreads();
  if (tid < 64) {
    float acc = 0.f;
    for (int k = 0; k < SEQL; ++k)
      acc += sc[k] * qkv[base + (size_t)k * 1536 + 1024 + hh * 64 + tid];
    o[((size_t)b * SEQL + qi) * DMODEL + hh * 64 + tid] = acc * inv;
  }
}

// MoE gate: softmax over 8 logits, top-2, renormalized weights -> wi[NTOK][8]
__global__ __launch_bounds__(256)
void gate_topk_kernel(const float* __restrict__ logits, float* __restrict__ wi)
{
  const int row = blockIdx.x * 256 + threadIdx.x;
  if (row >= NTOK) return;
  float p[NEXP]; float mx = -1e30f;
  #pragma unroll
  for (int e = 0; e < NEXP; ++e) { p[e] = logits[(size_t)row * NEXP + e]; mx = fmaxf(mx, p[e]); }
  float ssum = 0.f;
  #pragma unroll
  for (int e = 0; e < NEXP; ++e) { p[e] = __expf(p[e] - mx); ssum += p[e]; }
  #pragma unroll
  for (int e = 0; e < NEXP; ++e) p[e] /= ssum;
  int i1 = 0;
  #pragma unroll
  for (int e = 1; e < NEXP; ++e) if (p[e] > p[i1]) i1 = e;
  int i2 = (i1 == 0) ? 1 : 0;
  #pragma unroll
  for (int e = 0; e < NEXP; ++e) if (e != i1 && p[e] > p[i2]) i2 = e;
  const float t = p[i1] + p[i2];
  #pragma unroll
  for (int e = 0; e < NEXP; ++e) wi[(size_t)row * NEXP + e] = 0.f;
  wi[(size_t)row * NEXP + i1] = p[i1] / t;
  wi[(size_t)row * NEXP + i2] = p[i2] / t;
}

__global__ __launch_bounds__(256)
void add_kernel(float* __restrict__ x, const float* __restrict__ y, size_t n)
{
  const size_t i = (size_t)blockIdx.x * 256 + threadIdx.x;
  if (i < n) x[i] += y[i];
}

__global__ __launch_bounds__(256)
void beat_kernel(const int* __restrict__ tids, float* __restrict__ beat)
{
  const int i = blockIdx.x * 256 + threadIdx.x;
  if (i < NTOK) beat[i] = (tids[i] == 1) ? 1.f : 0.f;
}

// ---------------------------------------------------------------------------
extern "C" void kernel_launch(void* const* d_in, const int* in_sizes, int n_in,
                              void* d_out, int out_size, void* d_ws, size_t ws_size,
                              hipStream_t stream) {
  (void)in_sizes; (void)n_in; (void)out_size; (void)ws_size;
  int ci = 0;
  auto nf = [&]() { return (const float*)d_in[ci++]; };

  const float* seq  = nf();
  const int*   tids = (const int*)d_in[ci++];

  // ---- emb params ----
  const float *e1w1 = nf(), *e1b1 = nf(), *e1w2 = nf(), *e1b2 = nf();
  const float *e2w1 = nf(), *e2b1 = nf(), *e2w2 = nf(), *e2b2 = nf();
  const float *etype = nf(), *epos = nf(), *elnw = nf(), *elnb = nf();

  // ---- workspace ----
  float* ws = (float*)d_ws;
  size_t o = 0;
  float* x   = ws + o; o += (size_t)NTOK * DMODEL;
  float* xn  = ws + o; o += (size_t)NTOK * DMODEL;
  float* t0  = ws + o; o += (size_t)NTOK * DIP;
  float* t1  = ws + o; o += (size_t)NTOK * FFN;
  float* t2  = ws + o; o += (size_t)NTOK * DINNER;
  float* t4  = ws + o; o += (size_t)NTOK * DMODEL;
  float* t5  = ws + o; o += (size_t)NTOK * DMODEL;
  float* gl  = ws + o; o += (size_t)NTOK * NEXP;
  float* wi  = ws + o; o += (size_t)NTOK * NEXP;
  float* dtb = ws + o; o += (size_t)NTOK * HSSM;
  float* dab = ws + o; o += (size_t)NTOK * HSSM;
  float* beat= ws + o; o += (size_t)NTOK;

  auto gemm = [&](const float* A, int lda, const float* W, const float* bias,
                  const float* rs, int rss, float* Cp, int ldc,
                  int M, int N, int K, int act, int accum) {
    dim3 g((N + 63) / 64, (M + 127) / 128);
    gemm_wmma_kernel<<<g, 256, 0, stream>>>(A, lda, W, bias, rs, rss, Cp, ldc, M, N, K, act, accum);
  };

  // ---- embed ----
  gemm(seq, D2C, e1w1, e1b1, nullptr, 0, t1, DMODEL, NTOK, DMODEL, D1C, 1, 0);
  gemm(t1, DMODEL, e1w2, e1b2, nullptr, 0, t4, DMODEL, NTOK, DMODEL, DMODEL, 0, 0);
  gemm(seq, D2C, e2w1, e2b1, nullptr, 0, t1, DMODEL, NTOK, DMODEL, D2C, 1, 0);
  gemm(t1, DMODEL, e2w2, e2b2, nullptr, 0, t5, DMODEL, NTOK, DMODEL, DMODEL, 0, 0);
  embed_ln_kernel<<<NTOK, 256, 0, stream>>>(t4, t5, tids, etype, epos, elnw, elnb, x);

  // ---- blocks ----
  for (int blk = 0; blk < 7; ++blk) {
    const bool is_attn = (blk == 3);
    const size_t nmodel = (size_t)NTOK * DMODEL;
    if (!is_attn) {
      const float *n1 = nf(), *in_proj = nf(), *conv_w = nf(), *conv_b = nf();
      const float *dt_bias = nf(), *A_log = nf(), *Dp = nf(), *norm_w = nf();
      const float *out_proj = nf(), *n2 = nf();
      const float* gate = nf();
      const float *ew1[NEXP], *eb1[NEXP], *ew2[NEXP], *eb2[NEXP];
      for (int e = 0; e < NEXP; ++e) { ew1[e] = nf(); eb1[e] = nf(); ew2[e] = nf(); eb2[e] = nf(); }

      // mamba2
      rms512_kernel<<<NTOK, 256, 0, stream>>>(x, n1, xn);
      gemm(xn, DMODEL, in_proj, nullptr, nullptr, 0, t0, DIP, NTOK, DIP, DMODEL, 0, 0);
      {
        size_t tot = (size_t)NTOK * CONVDIM;
        conv_silu_kernel<<<(int)((tot + 255) / 256), 256, 0, stream>>>(t0, conv_w, conv_b, t1);
      }
      dt_kernel<<<(NTOK * HSSM + 255) / 256, 256, 0, stream>>>(t0, dt_bias, A_log, dtb, dab);
      scan_kernel<<<BSZ * HSSM, 256, 0, stream>>>(t1, dtb, dab, Dp, t2);
      gated_rms_kernel<<<NTOK, 256, 0, stream>>>(t2, t0, norm_w, t2);
      gemm(t2, DINNER, out_proj, nullptr, nullptr, 0, t4, DMODEL, NTOK, DMODEL, DINNER, 0, 0);
      add_kernel<<<(int)((nmodel + 255) / 256), 256, 0, stream>>>(x, t4, nmodel);

      // moe (dense all-experts; wi==0 for unselected -> exact)
      rms512_kernel<<<NTOK, 256, 0, stream>>>(x, n2, xn);
      gemm(xn, DMODEL, gate, nullptr, nullptr, 0, gl, NEXP, NTOK, NEXP, DMODEL, 0, 0);
      gate_topk_kernel<<<(NTOK + 255) / 256, 256, 0, stream>>>(gl, wi);
      for (int e = 0; e < NEXP; ++e) {
        gemm(xn, DMODEL, ew1[e], eb1[e], nullptr, 0, t1, FFN, NTOK, FFN, DMODEL, 1, 0);
        gemm(t1, FFN, ew2[e], eb2[e], wi + e, NEXP, t4, DMODEL, NTOK, DMODEL, FFN, 0, (e > 0) ? 1 : 0);
      }
      add_kernel<<<(int)((nmodel + 255) / 256), 256, 0, stream>>>(x, t4, nmodel);
    } else {
      const float *n1 = nf(), *wqkv = nf(), *bqkv = nf(), *wo = nf(), *bo = nf(), *n2 = nf();
      const float* gate = nf();
      const float *ew1[NEXP], *eb1[NEXP], *ew2[NEXP], *eb2[NEXP];
      for (int e = 0; e < NEXP; ++e) { ew1[e] = nf(); eb1[e] = nf(); ew2[e] = nf(); eb2[e] = nf(); }

      rms512_kernel<<<NTOK, 256, 0, stream>>>(x, n1, xn);
      gemm(xn, DMODEL, wqkv, bqkv, nullptr, 0, t0, 3 * DMODEL, NTOK, 3 * DMODEL, DMODEL, 0, 0);
      {
        dim3 g(SEQL, NHEAD, BSZ);
        attn_kernel<<<g, 256, 0, stream>>>(t0, tids, t4);
      }
      gemm(t4, DMODEL, wo, bo, nullptr, 0, t5, DMODEL, NTOK, DMODEL, DMODEL, 0, 0);
      add_kernel<<<(int)((nmodel + 255) / 256), 256, 0, stream>>>(x, t5, nmodel);

      rms512_kernel<<<NTOK, 256, 0, stream>>>(x, n2, xn);
      gemm(xn, DMODEL, gate, nullptr, nullptr, 0, gl, NEXP, NTOK, NEXP, DMODEL, 0, 0);
      gate_topk_kernel<<<(NTOK + 255) / 256, 256, 0, stream>>>(gl, wi);
      for (int e = 0; e < NEXP; ++e) {
        gemm(xn, DMODEL, ew1[e], eb1[e], nullptr, 0, t1, FFN, NTOK, FFN, DMODEL, 1, 0);
        gemm(t1, FFN, ew2[e], eb2[e], wi + e, NEXP, t4, DMODEL, NTOK, DMODEL, FFN, 0, (e > 0) ? 1 : 0);
      }
      add_kernel<<<(int)((nmodel + 255) / 256), 256, 0, stream>>>(x, t4, nmodel);
    }
  }

  // ---- final ln + heads ----
  const float *flnw = nf(), *flnb = nf();
  ln512_kernel<<<NTOK, 256, 0, stream>>>(x, flnw, flnb, xn);
  beat_kernel<<<(NTOK + 255) / 256, 256, 0, stream>>>(tids, beat);

  float* out = (float*)d_out;
  for (int hx = 0; hx < 6; ++hx) {            // fine(on,off,fr), corr(on,off,fr)
    const float* hw = nf();
    const float* hb = nf();
    gemm(xn, DMODEL, hw, hb, beat, 1, out + (size_t)hx * NTOK * NPITCH, NPITCH,
         NTOK, NPITCH, DMODEL, 3, 0);
  }
}
